// AttentionBlock_20435454394571
// MI455X (gfx1250) — matrix operand order repaired and checked by reading, not compiled, not run on hardware
//
#include <hip/hip_runtime.h>
#include <hip/hip_bf16.h>
#include <stdint.h>

// ---------------- fixed problem shape ----------------
#define BB   4
#define SEQ  2048
#define DIM  1024
#define HEADS 16
#define DH   64
#define BS   (BB * SEQ)          // 8192 rows

typedef __attribute__((ext_vector_type(16))) __bf16 v16bf;
typedef __attribute__((ext_vector_type(8)))  __bf16 v8bf;
typedef __attribute__((ext_vector_type(4)))  __bf16 v4bf;
typedef __attribute__((ext_vector_type(8)))  float  v8f;
typedef __attribute__((ext_vector_type(4)))  unsigned int u32x4;
typedef __attribute__((ext_vector_type(8)))  int i32x8;
typedef __attribute__((ext_vector_type(4)))  int i32x4;

// ---------------- workspace layout (bytes) ----------------
#define WS_XBF   ((size_t)0)                         // 16 MiB  x bf16 [BS,D]
#define WS_WQ    ((size_t)16u * 1024 * 1024)         // 2 MiB   wq bf16 [D,D]
#define WS_WK    (WS_WQ + (size_t)2 * 1024 * 1024)
#define WS_WV    (WS_WK + (size_t)2 * 1024 * 1024)
#define WS_W1    (WS_WV + (size_t)2 * 1024 * 1024)
#define WS_QBF   (WS_W1 + (size_t)2 * 1024 * 1024)   // 16 MiB  Q bf16 [B,H,S,dh]
#define WS_KBF   (WS_QBF + (size_t)16 * 1024 * 1024) // 16 MiB  K bf16 [B,H,S,dh]
#define WS_VT    (WS_KBF + (size_t)16 * 1024 * 1024) // 16 MiB  V^T bf16 [B,H,dh,S]
#define WS_ATTN  (WS_VT + (size_t)16 * 1024 * 1024)  // 32 MiB  attn f32 [BS,D] (reused as y)
#define WS_H     (WS_ATTN + (size_t)32 * 1024 * 1024)// 32 MiB  h f32 [BS,D]
#define WS_HBF   (WS_H + (size_t)32 * 1024 * 1024)   // 16 MiB  h bf16 [BS,D]

// ---------------- helpers ----------------
__device__ __forceinline__ v8f wmma_bf16(v16bf a, v16bf b, v8f c) {
  return __builtin_amdgcn_wmma_f32_16x16x32_bf16(false, a, false, b, (short)0, c,
                                                 false, false);
}

__device__ __forceinline__ v8f zero8() {
  v8f z;
#pragma unroll
  for (int i = 0; i < 8; ++i) z[i] = 0.0f;
  return z;
}

// A-matrix 16x32 bf16 fragment: lane<16 holds K {0..15}, lane>=16 holds
// K {16..31} of row (lane&15).  rowptr points at (row, kbase).
__device__ __forceinline__ v16bf load_a_frag(const __bf16* rowptr, int half) {
  return *(const v16bf*)(rowptr + 16 * half);
}

// B-matrix 32x16 bf16 fragment: lane holds column (lane&15), K in
// [16*half, 16*half+15] contiguous -> one 32B load.
__device__ __forceinline__ v16bf load_b_frag(const __bf16* colptr, int half) {
  return *(const v16bf*)(colptr + 16 * half);
}

// ---- Tensor Data Mover: DMA a [rows=64, cols=1024] bf16 panel (row stride
// 1024 elements, fully contiguous) from global into LDS.  D# per ISA ch.8.
// This toolchain exposes the 6-arg builtin:
//   (uint32x4 g0, int32x8 g1, int32x4 g2, int32x4 g3, int32x8 g4, i32 cpol)
__device__ __forceinline__ void tdm_load_w_panel(const __bf16* gsrc, __bf16* ldsdst) {
  const unsigned lds_off = (unsigned)(uintptr_t)ldsdst;          // LDS aperture low bits
  const unsigned long long ga = (unsigned long long)(uintptr_t)gsrc;
  u32x4 g0;
  g0[0] = 1u;                                        // count=1, user descriptor
  g0[1] = lds_off;                                   // lds_addr (bytes)
  g0[2] = (unsigned)(ga & 0xFFFFFFFFu);              // global_addr[31:0]
  g0[3] = (unsigned)((ga >> 32) & 0x1FFFFFFu) | (2u << 30);  // addr[56:32], type=2
  i32x8 g1;
  g1[0] = 0x00010000;            // wg_mask=0, data_size=1 (2 bytes)
  g1[1] = (int)(1024u << 16);    // tensor_dim0[15:0]=1024 (bits 63:48)
  g1[2] = (int)(64u << 16);      // tensor_dim0 hi=0, tensor_dim1=64
  g1[3] = (int)(1024u << 16);    // tensor_dim1 hi=0, tile_dim0=1024
  g1[4] = 64;                    // tile_dim1=64, tile_dim2=0
  g1[5] = 1024;                  // tensor_dim0_stride[31:0] = 1024 elements
  g1[6] = 0;                     // stride0 hi, tensor_dim1_stride lo
  g1[7] = 0;                     // tensor_dim1_stride hi
  i32x4 z4;
  z4[0] = 0; z4[1] = 0; z4[2] = 0; z4[3] = 0;        // 2-D tensor: groups 2/3 unused
  i32x8 z8;
#pragma unroll
  for (int i = 0; i < 8; ++i) z8[i] = 0;
  __builtin_amdgcn_tensor_load_to_lds(g0, g1, z4, z4, z8, 0);
}

// ---------------- f32 -> bf16 convert (4 elems/thread) ----------------
__global__ __launch_bounds__(256) void f32_to_bf16_k(const float* __restrict__ s,
                                                     __bf16* __restrict__ d, int n4) {
  int i = blockIdx.x * 256 + threadIdx.x;
  if (i < n4) {
    float4 v = ((const float4*)s)[i];
    v4bf o;
    o[0] = (__bf16)v.x; o[1] = (__bf16)v.y; o[2] = (__bf16)v.z; o[3] = (__bf16)v.w;
    ((v4bf*)d)[i] = o;
  }
}

// ---------------- WMMA GEMM: out = A[BS,D] @ W[D,D]^T + bias ----------------
// Weight panel for this block's 64 output columns (64 x 1024 bf16 = 128 KiB)
// is DMA'd once into LDS by the Tensor Data Mover, then the whole K loop
// reads B-fragments from LDS.  Dynamic LDS: 131072 bytes.
// mode 0: bf16 out, head-major  [B,H,S,dh]
// mode 1: bf16 out, transposed  [B,H,dh,S]   (for V^T)
// mode 2: f32  out, row-major   [BS,D]
__global__ __launch_bounds__(256) void gemm_wmma(const __bf16* __restrict__ A,
                                                 const __bf16* __restrict__ W,
                                                 const float* __restrict__ bias,
                                                 void* __restrict__ out, int mode) {
  extern __shared__ __bf16 ldsw[];        // [64][1024] bf16 weight panel

  const int lane = threadIdx.x & 31;
  const int w    = threadIdx.x >> 5;      // wave 0..7
  const int half = lane >> 4;
  const int lr   = lane & 15;
  const int m0 = blockIdx.x * 128 + w * 16;  // 16 rows per wave
  const int n0 = blockIdx.y * 64;            // 64 cols per wave (shared by block)

  // one wave issues the panel DMA; everyone waits at the barrier
  if (w == 0) {
    tdm_load_w_panel(W + (size_t)n0 * DIM, ldsw);
    __builtin_amdgcn_s_wait_tensorcnt((short)0);
  }
  __syncthreads();

  v8f acc[4] = {zero8(), zero8(), zero8(), zero8()};
  const __bf16* arow = A + (size_t)(m0 + lr) * DIM;

  for (int k = 0; k < DIM; k += 32) {
    v16bf af = load_a_frag(arow + k, half);
#pragma unroll
    for (int j = 0; j < 4; ++j) {
      const __bf16* wcol = ldsw + (size_t)(16 * j + lr) * DIM + k;  // LDS B-frag
      v16bf bf = load_b_frag(wcol, half);
      acc[j] = wmma_bf16(af, bf, acc[j]);
    }
  }

#pragma unroll
  for (int j = 0; j < 4; ++j) {
    const int n = n0 + 16 * j + lr;
    const float bv = bias[n];
#pragma unroll
    for (int r = 0; r < 8; ++r) {
      const int m = m0 + r + 8 * half;
      const float o = acc[j][r] + bv;
      if (mode == 2) {
        ((float*)out)[(size_t)m * DIM + n] = o;
      } else {
        const int bb = m >> 11, ss = m & (SEQ - 1);
        const int hh = n >> 6,  dd = n & (DH - 1);
        size_t idx;
        if (mode == 0) idx = (((size_t)bb * HEADS + hh) * SEQ + ss) * DH + dd;
        else           idx = (((size_t)bb * HEADS + hh) * DH + dd) * SEQ + ss;
        ((__bf16*)out)[idx] = (__bf16)o;
      }
    }
  }
}

// ---------------- flash attention (causal), one wave per 16-row q-tile ------
__global__ __launch_bounds__(256) void flash_attn(const __bf16* __restrict__ Q,
                                                  const __bf16* __restrict__ K,
                                                  const __bf16* __restrict__ VT,
                                                  float* __restrict__ O) {
  __shared__ __bf16 pbuf[8 * 16 * 32];  // 1 KiB per wave: P tile row-major 16x32

  const int lane = threadIdx.x & 31;
  const int w    = threadIdx.x >> 5;
  const int half = lane >> 4;
  const int lr   = lane & 15;

  const int tile = blockIdx.x * 8 + w;      // B*H*(S/16) = 8192 tiles
  const int qt = tile & 127;                // S/16 = 128
  const int h  = (tile >> 7) & (HEADS - 1);
  const int b  = tile >> 11;

  const size_t bh = (size_t)b * HEADS + h;
  const __bf16* qp = Q + (bh * SEQ + qt * 16 + lr) * DH;
  const __bf16* kp = K + bh * SEQ * DH;
  const __bf16* vp = VT + bh * DH * SEQ;
  __bf16* pw = pbuf + w * (16 * 32);

  const v16bf qa0 = load_a_frag(qp + 0,  half);
  const v16bf qa1 = load_a_frag(qp + 32, half);

  v8f acc[4] = {zero8(), zero8(), zero8(), zero8()};
  float mrow[8], lrow[8];
#pragma unroll
  for (int r = 0; r < 8; ++r) { mrow[r] = -1e30f; lrow[r] = 0.0f; }

  const int qmax = qt * 16 + 15;
  const int nblocks = (qmax >> 5) + 1;      // key blocks of 32

  for (int kb = 0; kb < nblocks; ++kb) {
    const int key0 = kb * 32;

    // prefetch next key block's K rows and V columns into cache
    if (kb + 1 < nblocks) {
      __builtin_prefetch(kp + (size_t)(key0 + 32 + lr) * DH, 0, 3);
      __builtin_prefetch(vp + (size_t)lr * SEQ + key0 + 32, 0, 3);
    }

    // --- scores: two 16x16 tiles, dh=64 -> 2 WMMA each ---
    v8f s0 = zero8(), s1 = zero8();
    {
      const __bf16* k0r = kp + (size_t)(key0 + lr) * DH;
      s0 = wmma_bf16(qa0, load_b_frag(k0r, half), s0);
      s0 = wmma_bf16(qa1, load_b_frag(k0r + 32, half), s0);
      const __bf16* k1r = kp + (size_t)(key0 + 16 + lr) * DH;
      s1 = wmma_bf16(qa0, load_b_frag(k1r, half), s1);
      s1 = wmma_bf16(qa1, load_b_frag(k1r + 32, half), s1);
    }

    // --- scale + causal mask + online softmax per row ---
#pragma unroll
    for (int r = 0; r < 8; ++r) {
      const int qrow = qt * 16 + r + 8 * half;
      float v0 = s0[r] * 0.125f;            // 1/sqrt(64)
      float v1 = s1[r] * 0.125f;
      if (key0 + lr > qrow)      v0 = -1e30f;
      if (key0 + 16 + lr > qrow) v1 = -1e30f;

      float mx = fmaxf(v0, v1);
      mx = fmaxf(mx, __shfl_xor(mx, 1, 16));
      mx = fmaxf(mx, __shfl_xor(mx, 2, 16));
      mx = fmaxf(mx, __shfl_xor(mx, 4, 16));
      mx = fmaxf(mx, __shfl_xor(mx, 8, 16));

      const float mnew = fmaxf(mrow[r], mx);
      const float p0 = __expf(v0 - mnew);
      const float p1 = __expf(v1 - mnew);
      float ps = p0 + p1;
      ps += __shfl_xor(ps, 1, 16);
      ps += __shfl_xor(ps, 2, 16);
      ps += __shfl_xor(ps, 4, 16);
      ps += __shfl_xor(ps, 8, 16);

      const float alpha = __expf(mrow[r] - mnew);
      lrow[r] = lrow[r] * alpha + ps;
      mrow[r] = mnew;
#pragma unroll
      for (int j = 0; j < 4; ++j) acc[j][r] *= alpha;

      const int prow = r + 8 * half;        // C-layout row -> row-major LDS
      pw[prow * 32 + lr]      = (__bf16)p0;
      pw[prow * 32 + 16 + lr] = (__bf16)p1;
    }

    // stores -> loads through LDS (same wave): keep order + wait DS
    asm volatile("s_wait_dscnt 0" ::: "memory");

    // --- repack P as A-fragment, then P(16x32) x V(32x64) ---
    const v16bf pf = load_a_frag(pw + lr * 32, half);
#pragma unroll
    for (int j = 0; j < 4; ++j) {
      const __bf16* vr = vp + (size_t)(16 * j + lr) * SEQ + key0;
      acc[j] = wmma_bf16(pf, load_b_frag(vr, half), acc[j]);
    }
  }

  // --- epilogue: normalize and store f32 attn[B,S,D] ---
#pragma unroll
  for (int r = 0; r < 8; ++r) {
    const float inv = 1.0f / lrow[r];
    const size_t rowoff = ((size_t)b * SEQ + qt * 16 + r + 8 * half) * DIM + h * DH;
#pragma unroll
    for (int j = 0; j < 4; ++j)
      O[rowoff + 16 * j + lr] = acc[j][r] * inv;
  }
}

// ---------------- fused residual + LayerNorm ----------------
// out = LN(a + c) * g + beta ; optionally also bf16 copy for next GEMM
__global__ __launch_bounds__(256) void add_ln(const float* __restrict__ a,
                                              const float* __restrict__ c,
                                              const float* __restrict__ g,
                                              const float* __restrict__ beta,
                                              float* __restrict__ out,
                                              __bf16* __restrict__ out_bf) {
  const int row = blockIdx.x;
  const int t = threadIdx.x;
  const float* ar = a + (size_t)row * DIM;
  const float* cr = c + (size_t)row * DIM;

  float e[4], s = 0.0f, sq = 0.0f;
#pragma unroll
  for (int i = 0; i < 4; ++i) {
    const int col = t + 256 * i;
    const float v = ar[col] + cr[col];
    e[i] = v; s += v; sq += v * v;
  }

  __shared__ float rs[256], rq[256];
  rs[t] = s; rq[t] = sq;
  __syncthreads();
  for (int off = 128; off > 0; off >>= 1) {
    if (t < off) { rs[t] += rs[t + off]; rq[t] += rq[t + off]; }
    __syncthreads();
  }
  const float mu = rs[0] * (1.0f / DIM);
  const float var = rq[0] * (1.0f / DIM) - mu * mu;
  const float rstd = rsqrtf(var + 1e-5f);

#pragma unroll
  for (int i = 0; i < 4; ++i) {
    const int col = t + 256 * i;
    const float o = (e[i] - mu) * rstd * g[col] + beta[col];
    out[(size_t)row * DIM + col] = o;
    if (out_bf) out_bf[(size_t)row * DIM + col] = (__bf16)o;
  }
}

// ---------------- host launch ----------------
extern "C" void kernel_launch(void* const* d_in, const int* in_sizes, int n_in,
                              void* d_out, int out_size, void* d_ws, size_t ws_size,
                              hipStream_t stream) {
  (void)in_sizes; (void)n_in; (void)out_size; (void)ws_size;
  const float* x      = (const float*)d_in[0];
  const float* wq_w   = (const float*)d_in[1];
  const float* wq_b   = (const float*)d_in[2];
  const float* wk_w   = (const float*)d_in[3];
  const float* wk_b   = (const float*)d_in[4];
  const float* wv_w   = (const float*)d_in[5];
  const float* wv_b   = (const float*)d_in[6];
  const float* ln1_g  = (const float*)d_in[7];
  const float* ln1_b  = (const float*)d_in[8];
  const float* lin1_w = (const float*)d_in[9];
  const float* lin1_b = (const float*)d_in[10];
  const float* ln2_g  = (const float*)d_in[11];
  const float* ln2_b  = (const float*)d_in[12];

  char* ws = (char*)d_ws;
  __bf16* xbf  = (__bf16*)(ws + WS_XBF);
  __bf16* wqb  = (__bf16*)(ws + WS_WQ);
  __bf16* wkb  = (__bf16*)(ws + WS_WK);
  __bf16* wvb  = (__bf16*)(ws + WS_WV);
  __bf16* w1b  = (__bf16*)(ws + WS_W1);
  __bf16* qbf  = (__bf16*)(ws + WS_QBF);
  __bf16* kbf  = (__bf16*)(ws + WS_KBF);
  __bf16* vtb  = (__bf16*)(ws + WS_VT);
  float*  attn = (float*)(ws + WS_ATTN);   // later reused as y
  float*  hbuf = (float*)(ws + WS_H);
  __bf16* hbf  = (__bf16*)(ws + WS_HBF);

  // 1) conversions
  {
    const int nx4 = BS * DIM / 4;
    f32_to_bf16_k<<<nx4 / 256, 256, 0, stream>>>(x, xbf, nx4);
    const int nw4 = DIM * DIM / 4;
    f32_to_bf16_k<<<nw4 / 256, 256, 0, stream>>>(wq_w, wqb, nw4);
    f32_to_bf16_k<<<nw4 / 256, 256, 0, stream>>>(wk_w, wkb, nw4);
    f32_to_bf16_k<<<nw4 / 256, 256, 0, stream>>>(wv_w, wvb, nw4);
    f32_to_bf16_k<<<nw4 / 256, 256, 0, stream>>>(lin1_w, w1b, nw4);
  }

  const size_t lds_bytes = 64 * DIM * sizeof(__bf16);   // 128 KiB weight panel

  // 2) QKV projections (WMMA + TDM-staged weights)
  dim3 gg(BS / 128, DIM / 64);
  gemm_wmma<<<gg, 256, lds_bytes, stream>>>(xbf, wqb, wq_b, qbf, 0);
  gemm_wmma<<<gg, 256, lds_bytes, stream>>>(xbf, wkb, wk_b, kbf, 0);
  gemm_wmma<<<gg, 256, lds_bytes, stream>>>(xbf, wvb, wv_b, vtb, 1);

  // 3) causal flash attention
  flash_attn<<<(BB * HEADS * (SEQ / 16)) / 8, 256, 0, stream>>>(qbf, kbf, vtb, attn);

  // 4) h = LN(x + attn)
  add_ln<<<BS, 256, 0, stream>>>(x, attn, ln1_g, ln1_b, hbuf, hbf);

  // 5) y = h @ lin1^T + b (WMMA, f32 out into attn buffer)
  gemm_wmma<<<gg, 256, lds_bytes, stream>>>(hbf, w1b, lin1_b, attn, 2);

  // 6) out = LN(h + y)
  add_ln<<<BS, 256, 0, stream>>>(hbuf, attn, ln2_g, ln2_b, (float*)d_out, nullptr);
}
